// Net_10299331576450
// MI455X (gfx1250) — compile-verified, matrix operand order checked by reading, not run on hardware
//
#include <hip/hip_runtime.h>
#include <hip/hip_bf16.h>
#include <math.h>

typedef __attribute__((ext_vector_type(16))) _Float16 v16h;
typedef __attribute__((ext_vector_type(8)))  float    v8f;
typedef __attribute__((ext_vector_type(4)))  float    v4f;

#define F_IN 128
#define DIM  16
#define NCLS 10

// ---------------------------------------------------------------------------
// Degree / normalization kernels
// ---------------------------------------------------------------------------
__global__ void zero_f32_kernel(float* __restrict__ p, int n) {
    int i = blockIdx.x * blockDim.x + threadIdx.x;
    if (i < n) p[i] = 0.0f;
}

__global__ void deg_accum_kernel(const int* __restrict__ dst, float* __restrict__ deg, int nE) {
    int e = blockIdx.x * blockDim.x + threadIdx.x;
    if (e < nE) atomicAdd(&deg[dst[e]], 1.0f);
}

// dinv[i] = rsqrt(deg[i] + 1)   (+1 = self loop; deg always > 0)
__global__ void dinv_kernel(float* __restrict__ deg_dinv, int n) {
    int i = blockIdx.x * blockDim.x + threadIdx.x;
    if (i < n) {
        float d = deg_dinv[i] + 1.0f;
        deg_dinv[i] = (d > 0.0f) ? rsqrtf(d) : 0.0f;
    }
}

__global__ void norm_kernel(const int* __restrict__ src, const int* __restrict__ dst,
                            const float* __restrict__ dinv, float* __restrict__ norm, int nE) {
    int e = blockIdx.x * blockDim.x + threadIdx.x;
    if (e < nE) norm[e] = dinv[src[e]] * dinv[dst[e]];
}

// ---------------------------------------------------------------------------
// WMMA GEMM, K = 128 (layer 1):  OUT[N x 16] = X[N x 128] @ W[128 x 16]
// One wave per 16-row tile; v_wmma_f32_16x16x32_f16 chained over 4 K-chunks.
// A-operand per lane = two contiguous 8-float runs -> 4x global_load_b128.
//
// A layout (16-bit A 16x32, ISA 7.12.2): lane L holds row m = L%16,
//   element e -> k(e) = (L/16)*8 + e + (e>=8 ? 8 : 0)
// B is the symmetric column image: lane L holds column n = L%16, same k(e).
// D layout: VGPR r, lane L -> row (r + 8*(L/16)), col (L%16).
// ---------------------------------------------------------------------------
__global__ void gemm_k128_wmma(const float* __restrict__ X, const float* __restrict__ W,
                               float* __restrict__ OUT, int ntiles) {
    const int wid  = threadIdx.x >> 5;
    const int lane = threadIdx.x & 31;
    const int tile = blockIdx.x * (blockDim.x >> 5) + wid;
    if (tile >= ntiles) return;                 // wave-uniform: EXEC stays all-1s

    const int m     = lane & 15;
    const int half  = lane >> 4;
    const int kbase = half * 8;
    const int row   = tile * 16 + m;            // A row this lane supplies
    const int n     = m;                        // B column this lane supplies

    const float* Arow = X + (size_t)row * F_IN;

    v8f acc = {};
    #pragma unroll
    for (int c = 0; c < 4; ++c) {               // K chunks of 32
        // A: k = c*32 + kbase + [0..7]  and  c*32 + kbase + 16 + [0..7]
        const v4f* p0 = (const v4f*)(Arow + c * 32 + kbase);
        const v4f* p1 = (const v4f*)(Arow + c * 32 + kbase + 16);
        v4f x0 = p0[0], x1 = p0[1];
        v4f x2 = p1[0], x3 = p1[1];

        v16h a, b;
        #pragma unroll
        for (int e = 0; e < 4; ++e) {
            a[e]      = (_Float16)x0[e];
            a[e + 4]  = (_Float16)x1[e];
            a[e + 8]  = (_Float16)x2[e];
            a[e + 12] = (_Float16)x3[e];
        }
        #pragma unroll
        for (int e = 0; e < 16; ++e) {
            const int gk = c * 32 + kbase + e + ((e >= 8) ? 8 : 0);
            b[e] = (_Float16)W[gk * DIM + n];
        }
        acc = __builtin_amdgcn_wmma_f32_16x16x32_f16(
                  false, a, false, b, (short)0, acc, false, false);
    }
    #pragma unroll
    for (int r = 0; r < 8; ++r)
        OUT[(tile * 16 + half * 8 + r) * DIM + n] = acc[r];
}

// ---------------------------------------------------------------------------
// WMMA GEMM, K = 16 (layers 2,3):  OUT = relu?(H)[N x 16] @ W[16 x 16]
// K padded 16 -> 32 with zeros (one WMMA per tile); ReLU fused into A load.
// ---------------------------------------------------------------------------
__global__ void gemm_k16_wmma(const float* __restrict__ H, const float* __restrict__ W,
                              float* __restrict__ OUT, int ntiles, int do_relu) {
    const int wid  = threadIdx.x >> 5;
    const int lane = threadIdx.x & 31;
    const int tile = blockIdx.x * (blockDim.x >> 5) + wid;
    if (tile >= ntiles) return;

    const int m     = lane & 15;
    const int half  = lane >> 4;
    const int kbase = half * 8;
    const int row   = tile * 16 + m;
    const int n     = m;

    // A: 8 contiguous floats H[row, kbase..kbase+7] -> 2x b128
    const v4f* hp = (const v4f*)(H + (size_t)row * DIM + kbase);
    v4f h0 = hp[0], h1 = hp[1];

    v16h a = {};                                 // elements 8..15 (k>=16) stay 0
    v16h b = {};
    #pragma unroll
    for (int e = 0; e < 4; ++e) {
        float x0 = h0[e], x1 = h1[e];
        if (do_relu) { x0 = fmaxf(x0, 0.0f); x1 = fmaxf(x1, 0.0f); }
        a[e]     = (_Float16)x0;
        a[e + 4] = (_Float16)x1;
    }
    #pragma unroll
    for (int e = 0; e < 8; ++e)
        b[e] = (_Float16)W[(kbase + e) * DIM + n];

    v8f acc = {};
    acc = __builtin_amdgcn_wmma_f32_16x16x32_f16(
              false, a, false, b, (short)0, acc, false, false);
    #pragma unroll
    for (int r = 0; r < 8; ++r)
        OUT[(tile * 16 + half * 8 + r) * DIM + n] = acc[r];
}

// ---------------------------------------------------------------------------
// OUT[i, j0..j0+3] = bias[j0..] + HW[i, j0..] * dinv[i]^2   (bias + self loop)
// one thread per 4 consecutive features
// ---------------------------------------------------------------------------
__global__ void init_bias_selfloop(float* __restrict__ OUT, const float* __restrict__ HW,
                                   const float* __restrict__ dinv,
                                   const float* __restrict__ bias, int n) {
    int g = blockIdx.x * blockDim.x + threadIdx.x;      // n*4 groups
    if (g < n * (DIM / 4)) {
        int i  = g >> 2;
        int j0 = (g & 3) * 4;
        float di  = dinv[i];
        float di2 = di * di;
        v4f hw = ((const v4f*)HW)[g];
        v4f bb = *(const v4f*)(bias + j0);
        v4f o;
        #pragma unroll
        for (int t = 0; t < 4; ++t) o[t] = bb[t] + hw[t] * di2;
        ((v4f*)OUT)[g] = o;
    }
}

// ---------------------------------------------------------------------------
// Edge scatter-add: OUT[dst,j] += HW[src,j] * norm[e]   (16 lanes per edge)
// ---------------------------------------------------------------------------
__global__ void scatter_edges(const float* __restrict__ HW, const float* __restrict__ norm,
                              const int* __restrict__ src, const int* __restrict__ dst,
                              float* __restrict__ OUT, int nE) {
    long long gid = (long long)blockIdx.x * blockDim.x + threadIdx.x;
    long long e   = gid >> 4;
    int       j   = (int)(gid & 15);
    if (e >= nE) return;
    int s = src[e], d = dst[e];
    float v = HW[(long long)s * DIM + j] * norm[e];
    atomicAdd(&OUT[(long long)d * DIM + j], v);
}

// ---------------------------------------------------------------------------
// logits = H @ Wl + bl ; out = log_softmax(logits)
// ---------------------------------------------------------------------------
__global__ void logits_logsoftmax(const float* __restrict__ H, const float* __restrict__ Wl,
                                  const float* __restrict__ bl, float* __restrict__ out, int n) {
    int i = blockIdx.x * blockDim.x + threadIdx.x;
    if (i >= n) return;
    const v4f* hp = (const v4f*)(H + (size_t)i * DIM);
    v4f hv[4];
    #pragma unroll
    for (int q = 0; q < 4; ++q) hv[q] = hp[q];

    float l[NCLS];
    #pragma unroll
    for (int c = 0; c < NCLS; ++c) {
        float s = bl[c];
        #pragma unroll
        for (int q = 0; q < 4; ++q)
            #pragma unroll
            for (int t = 0; t < 4; ++t)
                s += hv[q][t] * Wl[(q * 4 + t) * NCLS + c];
        l[c] = s;
    }
    float mx = l[0];
    #pragma unroll
    for (int c = 1; c < NCLS; ++c) mx = fmaxf(mx, l[c]);
    float se = 0.0f;
    #pragma unroll
    for (int c = 0; c < NCLS; ++c) se += expf(l[c] - mx);
    float lse = mx + logf(se);
    #pragma unroll
    for (int c = 0; c < NCLS; ++c) out[i * NCLS + c] = l[c] - lse;
}

// ---------------------------------------------------------------------------
extern "C" void kernel_launch(void* const* d_in, const int* in_sizes, int n_in,
                              void* d_out, int out_size, void* d_ws, size_t ws_size,
                              hipStream_t stream) {
    const float* x  = (const float*)d_in[0];
    const int*   ei = (const int*)d_in[1];      // edge_index [2,E]
    const float* W1 = (const float*)d_in[2];
    const float* b1 = (const float*)d_in[3];
    const float* W2 = (const float*)d_in[4];
    const float* b2 = (const float*)d_in[5];
    const float* W3 = (const float*)d_in[6];
    const float* b3 = (const float*)d_in[7];
    const float* Wl = (const float*)d_in[8];
    const float* bl = (const float*)d_in[9];

    const int N = in_sizes[0] / F_IN;
    const int E = in_sizes[1] / 2;
    const int* src = ei;
    const int* dst = ei + E;

    // workspace layout (floats): dinv[N] | norm[E] | bufA[N*16] | bufB[N*16]
    float* ws   = (float*)d_ws;
    float* dinv = ws;
    float* nrm  = ws + N;
    float* bufA = nrm + E;
    float* bufB = bufA + (size_t)N * DIM;

    const int TB = 256;
    const int ntiles   = (N + 15) / 16;
    const int gemm_bl  = (ntiles + 7) / 8;                  // 8 waves / block
    const int nodes_bl = (N + TB - 1) / TB;
    const int edge_bl  = (E + TB - 1) / TB;
    const int nd4_bl   = (N * (DIM / 4) + TB - 1) / TB;
    const int scat_bl  = (int)(((long long)E * DIM + TB - 1) / TB);

    // ---- symmetric normalization ----
    zero_f32_kernel<<<nodes_bl, TB, 0, stream>>>(dinv, N);
    deg_accum_kernel<<<edge_bl, TB, 0, stream>>>(dst, dinv, E);
    dinv_kernel<<<nodes_bl, TB, 0, stream>>>(dinv, N);
    norm_kernel<<<edge_bl, TB, 0, stream>>>(src, dst, dinv, nrm, E);

    // ---- layer 1: X@W1 -> aggregate ----
    gemm_k128_wmma<<<gemm_bl, TB, 0, stream>>>(x, W1, bufA, ntiles);
    init_bias_selfloop<<<nd4_bl, TB, 0, stream>>>(bufB, bufA, dinv, b1, N);
    scatter_edges<<<scat_bl, TB, 0, stream>>>(bufA, nrm, src, dst, bufB, E);

    // ---- layer 2: relu(h1)@W2 -> aggregate ----
    gemm_k16_wmma<<<gemm_bl, TB, 0, stream>>>(bufB, W2, bufA, ntiles, 1);
    init_bias_selfloop<<<nd4_bl, TB, 0, stream>>>(bufB, bufA, dinv, b2, N);
    scatter_edges<<<scat_bl, TB, 0, stream>>>(bufA, nrm, src, dst, bufB, E);

    // ---- layer 3: relu(h2)@W3 -> aggregate (no relu after) ----
    gemm_k16_wmma<<<gemm_bl, TB, 0, stream>>>(bufB, W3, bufA, ntiles, 1);
    init_bias_selfloop<<<nd4_bl, TB, 0, stream>>>(bufB, bufA, dinv, b3, N);
    scatter_edges<<<scat_bl, TB, 0, stream>>>(bufA, nrm, src, dst, bufB, E);

    // ---- classifier + log_softmax ----
    logits_logsoftmax<<<nodes_bl, TB, 0, stream>>>(bufB, Wl, bl, (float*)d_out, N);
}
